// DPL_82935818486221
// MI455X (gfx1250) — matrix operand order repaired and checked by reading, not compiled, not run on hardware
//
#include <hip/hip_runtime.h>

// Problem constants (from reference)
#define B_TOT 256
#define L_SEQ 1024
#define KDIM  64
#define NPP   16
#define NAA   32
#define XDIM  112            // NP+NA+K
#define XPAD  128            // padded K-dim for 4 WMMA k-steps of 32
#define KB    4              // XPAD/32
#define HID   448            // 4*S_IN
#define KB2   14             // HID/32 k-steps for MLP layer 2
#define NCOL  5632           // 4096 (r) + 1024 (i) + 64 (o) + 448 (mlp1)
#define NT    352            // NCOL/16 column tiles
#define BT    16             // batch tile (WMMA M)
#define NBT   16             // number of batch tiles = B/BT
#define WP_ELEMS (KB*NT*32*16)        // 720896 bf16 elements
#define WP_BYTES (WP_ELEMS*2)
#define W2_ELEMS (KB2*4*32*16)        // 28672 bf16 elements (mlp_w2 fragments)
#define BIAS_BYTES (NCOL*4)

typedef __attribute__((ext_vector_type(16))) __bf16 v16bf;
typedef __attribute__((ext_vector_type(8)))  float  v8f;

// B-fragment packing index helpers (v_wmma_f32_16x16x32_bf16):
//   B frag: lane l holds col = base + (l&15), K = (slot>>3)*16 + (l>>4)*8 + (slot&7)  (slot 0..15)
//   A frag: lane l<16 holds row l,  K in {0..7,16..23}; lane l>=16 holds row l-16, K in {8..15,24..31}
//           slot(k) = (kr&7) | ((kr>>4)<<3) with kr = (k&31) - 8*h, h = ((k>>3)&1)

// ---------------------------------------------------------------------------
// Prepack: fused gate weight [112,5632] + mlp_w2 [448,64], both in WMMA
// B-fragment bf16 layout, plus the fused fp32 bias vector.
// ---------------------------------------------------------------------------
__global__ void DPL_prepack(
    const float* __restrict__ w_rp, const float* __restrict__ w_ra, const float* __restrict__ w_rs,
    const float* __restrict__ b_r,
    const float* __restrict__ w_ip, const float* __restrict__ w_ia, const float* __restrict__ w_is,
    const float* __restrict__ b_i,
    const float* __restrict__ w_op, const float* __restrict__ w_oa, const float* __restrict__ w_os,
    const float* __restrict__ b_o,
    const float* __restrict__ mlp_w1, const float* __restrict__ mlp_b1,
    const float* __restrict__ mlp_w2,
    unsigned short* __restrict__ wpack_u, float* __restrict__ biasws,
    unsigned short* __restrict__ w2pack_u)
{
    int idx = blockIdx.x * blockDim.x + threadIdx.x;
    if (idx < WP_ELEMS) {
        int slot  = idx & 15;
        int lane  = (idx >> 4) & 31;
        int ktile = idx >> 9;
        int nt = ktile % NT;
        int kb = ktile / NT;
        int col = nt * 16 + (lane & 15);
        int row = kb * 32 + ((slot >> 3) << 4) + (lane >> 4) * 8 + (slot & 7);   // row of x (0..127)
        float v = 0.f;
        if (row < XDIM) {
            if (col < 4096) {                                   // routing gate
                v = (row < NPP)       ? w_rp[row * 4096 + col]
                  : (row < NPP + NAA) ? w_ra[(row - NPP) * 4096 + col]
                                      : w_rs[(row - 48) * 4096 + col];
            } else if (col < 5120) {                            // input gate
                int c = col - 4096;
                v = (row < NPP) ? w_ip[row * 1024 + c]
                  : (row < 48)  ? w_ia[(row - NPP) * 1024 + c]
                                : w_is[(row - 48) * 1024 + c];
            } else if (col < 5184) {                            // output gate
                int c = col - 5120;
                v = (row < NPP) ? w_op[row * 64 + c]
                  : (row < 48)  ? w_oa[(row - NPP) * 64 + c]
                                : w_os[(row - 48) * 64 + c];
            } else {                                            // MLP layer 1
                v = mlp_w1[row * HID + (col - 5184)];
            }
        }
        ((__bf16*)wpack_u)[idx] = (__bf16)v;
    } else if (idx < WP_ELEMS + NCOL) {
        int col = idx - WP_ELEMS;
        float v = (col < 4096) ? b_r[col]
                : (col < 5120) ? b_i[col - 4096]
                : (col < 5184) ? b_o[col - 5120]
                               : mlp_b1[col - 5184];
        biasws[col] = v;
    } else if (idx < WP_ELEMS + NCOL + W2_ELEMS) {
        int i2 = idx - WP_ELEMS - NCOL;
        int slot  = i2 & 15;
        int lane  = (i2 >> 4) & 31;
        int ktile = i2 >> 9;
        int nt2 = ktile & 3;           // 4 column tiles (K=64 outputs)
        int kb2 = ktile >> 2;          // 14 k-blocks (HID=448)
        int col = nt2 * 16 + (lane & 15);
        int row = kb2 * 32 + ((slot >> 3) << 4) + (lane >> 4) * 8 + (slot & 7);    // 0..447, exact
        ((__bf16*)w2pack_u)[i2] = (__bf16)mlp_w2[row * KDIM + col];
    }
}

// ---------------------------------------------------------------------------
// 4-tile GEMM group: C[16 x 64] = relu(x @ W[:, group] + bias), then an
// in-register softmax across the 64 columns of each batch row.
// (C layout: VGPR j <-> rows j / j+8, N <-> lane&15.)
// ---------------------------------------------------------------------------
__device__ __forceinline__ void gemm_relu_group4(
    const v16bf afrag[KB], const __bf16* __restrict__ wp,
    int ntbase, int colbase, const float* __restrict__ biasws,
    int lane, float cf[4][8])
{
    const int lanecol = lane & 15;
    #pragma unroll
    for (int tt = 0; tt < 4; ++tt) {
        v8f c = {0.f, 0.f, 0.f, 0.f, 0.f, 0.f, 0.f, 0.f};
        #pragma unroll
        for (int kb = 0; kb < KB; ++kb) {
            v16bf bfrag = *(const v16bf*)&wp[(((kb * NT + ntbase + tt) * 32) + lane) * 16];
            c = __builtin_amdgcn_wmma_f32_16x16x32_bf16(false, afrag[kb], false, bfrag,
                                                        (short)0, c, false, false);
        }
        float bv = biasws[colbase + tt * 16 + lanecol];
        #pragma unroll
        for (int j = 0; j < 8; ++j) cf[tt][j] = fmaxf(c[j] + bv, 0.f);
    }
}

__device__ __forceinline__ void softmax_rows4(float cf[4][8])
{
    #pragma unroll
    for (int j = 0; j < 8; ++j) {
        float mx = fmaxf(fmaxf(cf[0][j], cf[1][j]), fmaxf(cf[2][j], cf[3][j]));
        #pragma unroll
        for (int m = 1; m < 16; m <<= 1) mx = fmaxf(mx, __shfl_xor(mx, m, 32));
        float sum = 0.f;
        #pragma unroll
        for (int tt = 0; tt < 4; ++tt) { float e = __expf(cf[tt][j] - mx); cf[tt][j] = e; sum += e; }
        #pragma unroll
        for (int m = 1; m < 16; m <<= 1) sum += __shfl_xor(sum, m, 32);
        float inv = 1.f / sum;
        #pragma unroll
        for (int tt = 0; tt < 4; ++tt) cf[tt][j] *= inv;
    }
}

// ---------------------------------------------------------------------------
// Persistent scan kernel: one workgroup per 16-row batch tile, 32 wave32s.
// All recurrent state and per-step intermediates are LDS-resident; packed
// weights (1.5 MB bf16) stay in the 192 MB L2 across all 1024 steps.
// ---------------------------------------------------------------------------
__global__ __launch_bounds__(1024, 1)
void DPL_scan(const float* __restrict__ prec, const float* __restrict__ attr,
              const float* __restrict__ s_max_p, const float* __restrict__ s_fc_p,
              const float* __restrict__ mlp_b2,
              const unsigned short* __restrict__ wpack_u, const float* __restrict__ biasws,
              const unsigned short* __restrict__ w2pack_u,
              float* __restrict__ s_out, float* __restrict__ q_out)
{
    __shared__ float s_cur[BT][KDIM];                        // recurrent state
    __shared__ __align__(32) __bf16 x_pack[KB][32][16];      // A fragments (gate GEMM)
    __shared__ __align__(32) __bf16 hid2_pack[KB2][32][16];  // A fragments (MLP layer 2)
    __shared__ float inv_sum[BT];
    __shared__ float ptile[BT][NPP];
    __shared__ float s_re_l[BT][KDIM];
    __shared__ float m_slab[NPP][BT][KDIM];                  // per-m input-gate slabs
    __shared__ float o1_l[BT][KDIM];
    __shared__ float n_ci_l[BT][KDIM];
    __shared__ float qpart[BT][KDIM];

    const __bf16* wp  = (const __bf16*)wpack_u;
    const __bf16* w2p = (const __bf16*)w2pack_u;
    const int tid      = threadIdx.x;
    const int lane     = tid & 31;
    const int wave     = tid >> 5;
    const int lanecol  = lane & 15;
    const int lanehalf = lane >> 4;
    const int bg       = blockIdx.x;

    ((float*)s_cur)[tid] = 0.f;     // s0 = 0 (BT*KDIM == 1024)
    __syncthreads();

    for (int t = 0; t < L_SEQ; ++t) {
        // Compiler fence: keep the weight-fragment loads inside the loop so
        // they stay cheap L2-resident global_load_b128 instead of being
        // LICM-hoisted and spilled to scratch.
        asm volatile("" ::: "memory");

        // ---- phase 1a: h_s normalizer ----
        if (tid < BT) {
            float s = 0.f;
            #pragma unroll 8
            for (int k = 0; k < KDIM; ++k) s += s_cur[tid][k];
            inv_sum[tid] = 1.f / (s + 1e-8f);
        }
        __syncthreads();

        // ---- phase 1b: build bf16 A fragments for x=[p,a,h_s,0-pad] ----
        #pragma unroll
        for (int e0 = 0; e0 < 2; ++e0) {
            int e = e0 * 1024 + tid;            // (b, kglobal) over 16*128
            int b = e >> 7, kg = e & 127;
            float v = 0.f;
            if (kg < NPP)           v = prec[((size_t)(bg * BT + b) * L_SEQ + t) * NPP + kg];
            else if (kg < NPP+NAA)  v = attr[((size_t)(bg * BT + b) * L_SEQ + t) * NAA + (kg - NPP)];
            else if (kg < XDIM)     v = s_cur[b][kg - 48] * inv_sum[b];
            int kb = kg >> 5, kk = kg & 31;
            int h  = (kk >> 3) & 1;
            int kr = kk - 8 * h;
            int slot = (kr & 7) | ((kr >> 4) << 3);
            x_pack[kb][b + 16 * h][slot] = (__bf16)v;
        }
        if (tid < BT * NPP) {
            int b = tid >> 4, n = tid & 15;
            ptile[b][n] = prec[((size_t)(bg * BT + b) * L_SEQ + t) * NPP + n];
        }
        if (t + 1 < L_SEQ && tid < BT) {   // warm next step's inputs (global_prefetch_b8)
            __builtin_prefetch(&prec[((size_t)(bg * BT + tid) * L_SEQ + (t + 1)) * NPP], 0, 0);
            __builtin_prefetch(&attr[((size_t)(bg * BT + tid) * L_SEQ + (t + 1)) * NAA], 0, 0);
        }
        __syncthreads();

        // ---- phase 2: fused WMMA gate GEMM + epilogues ----
        v16bf afrag[KB];
        #pragma unroll
        for (int kb = 0; kb < KB; ++kb)
            afrag[kb] = *(const v16bf*)&x_pack[kb][lane][0];

        // routing gate: each wave owns 2 k-rows (64 columns each)
        #pragma unroll
        for (int rr = 0; rr < 2; ++rr) {
            int krow = wave * 2 + rr;
            float cf[4][8];
            gemm_relu_group4(afrag, wp, krow * 4, krow * 64, biasws, lane, cf);
            softmax_rows4(cf);
            #pragma unroll
            for (int j = 0; j < 8; ++j) {       // s_re[b,krow] = r_s . s
                int b = j + 8 * lanehalf;
                float acc = 0.f;
                #pragma unroll
                for (int tt = 0; tt < 4; ++tt) acc += cf[tt][j] * s_cur[b][tt * 16 + lanecol];
                #pragma unroll
                for (int m = 1; m < 16; m <<= 1) acc += __shfl_xor(acc, m, 32);
                if (lanecol == 0) s_re_l[b][krow] = acc;
            }
        }

        if (wave < NPP) {
            // input gate slice m = wave: softmax over K, scale by p[b,m]
            float cf[4][8];
            gemm_relu_group4(afrag, wp, 256 + wave * 4, 4096 + wave * 64, biasws, lane, cf);
            softmax_rows4(cf);
            #pragma unroll
            for (int j = 0; j < 8; ++j) {
                int b = j + 8 * lanehalf;
                float pv = ptile[b][wave];
                #pragma unroll
                for (int tt = 0; tt < 4; ++tt)
                    m_slab[wave][b][tt * 16 + lanecol] = cf[tt][j] * pv;
            }
        } else if (wave == 16) {
            // output gate
            float cf[4][8];
            gemm_relu_group4(afrag, wp, 320, 5120, biasws, lane, cf);
            softmax_rows4(cf);
            #pragma unroll
            for (int j = 0; j < 8; ++j) {
                int b = j + 8 * lanehalf;
                #pragma unroll
                for (int tt = 0; tt < 4; ++tt)
                    o1_l[b][tt * 16 + lanecol] = cf[tt][j];
            }
        } else if (wave >= 17 && wave <= 30) {
            // MLP layer-1 tiles (2 per wave): relu, write hidden directly in
            // A-fragment layout for the layer-2 WMMA.
            #pragma unroll
            for (int u = 0; u < 2; ++u) {
                int mt = (wave - 17) * 2 + u;   // 0..27
                int nt = 324 + mt;
                v8f c = {0.f, 0.f, 0.f, 0.f, 0.f, 0.f, 0.f, 0.f};
                #pragma unroll
                for (int kb = 0; kb < KB; ++kb) {
                    v16bf bfrag = *(const v16bf*)&wp[(((kb * NT + nt) * 32) + lane) * 16];
                    c = __builtin_amdgcn_wmma_f32_16x16x32_bf16(false, afrag[kb], false, bfrag,
                                                                (short)0, c, false, false);
                }
                float bv = biasws[5184 + mt * 16 + lanecol];
                int hcol = mt * 16 + lanecol;
                int kb2 = hcol >> 5, kk = hcol & 31;
                int h   = (kk >> 3) & 1;
                int kr  = kk - 8 * h;
                int slot = (kr & 7) | ((kr >> 4) << 3);
                #pragma unroll
                for (int j = 0; j < 8; ++j) {
                    int b = j + 8 * lanehalf;
                    hid2_pack[kb2][b + 16 * h][slot] = (__bf16)fmaxf(c[j] + bv, 0.f);
                }
            }
        }
        __syncthreads();

        // ---- phase 2b: MLP layer 2 via WMMA (4 waves, one 16-col tile each) ----
        if (wave < 4) {
            v8f c = {0.f, 0.f, 0.f, 0.f, 0.f, 0.f, 0.f, 0.f};
            #pragma unroll
            for (int kb2 = 0; kb2 < KB2; ++kb2) {
                v16bf a2 = *(const v16bf*)&hid2_pack[kb2][lane][0];
                v16bf b2f = *(const v16bf*)&w2p[(((kb2 * 4 + wave) * 32) + lane) * 16];
                c = __builtin_amdgcn_wmma_f32_16x16x32_bf16(false, a2, false, b2f,
                                                            (short)0, c, false, false);
            }
            float bv = mlp_b2[wave * 16 + lanecol];
            #pragma unroll
            for (int j = 0; j < 8; ++j) {
                int b = j + 8 * lanehalf;
                n_ci_l[b][wave * 16 + lanecol] = fmaxf(c[j] + bv, 0.f);
            }
        }
        __syncthreads();

        // ---- phase 3: state update, one thread per (b,k) ----
        {
            int b = tid >> 6, k = tid & 63;
            float m_in = 0.f;
            #pragma unroll
            for (int m = 0; m < NPP; ++m) m_in += m_slab[m][b][k];   // fixed-order, deterministic
            float sre = s_re_l[b][k];
            float smx = s_max_p[k];
            float fcc = fminf(s_fc_p[k], smx);
            float q1  = m_in * (tanhf((sre - smx) * 4.f) + 1.f) * 0.5f;
            float q2  = o1_l[b][k] * fmaxf(sre - fcc, 0.f);
            float snew = s_cur[b][k] + m_in - q1 - q2 - n_ci_l[b][k];
            s_cur[b][k] = snew;
            s_out[((size_t)(bg * BT + b) * L_SEQ + t) * KDIM + k] = snew;
            qpart[b][k] = q1 + q2;
        }
        __syncthreads();
        if (tid < BT) {
            float q = 0.f;
            #pragma unroll 8
            for (int k = 0; k < KDIM; ++k) q += qpart[tid][k];
            q_out[(size_t)(bg * BT + tid) * L_SEQ + t] = q;
        }
    }
}

// ---------------------------------------------------------------------------
extern "C" void kernel_launch(void* const* d_in, const int* in_sizes, int n_in,
                              void* d_out, int out_size, void* d_ws, size_t ws_size,
                              hipStream_t stream)
{
    (void)in_sizes; (void)n_in; (void)out_size; (void)ws_size;
    const float* prec   = (const float*)d_in[0];
    const float* attr   = (const float*)d_in[1];
    const float* w_rp   = (const float*)d_in[2];
    const float* w_ra   = (const float*)d_in[3];
    const float* w_rs   = (const float*)d_in[4];
    const float* b_r    = (const float*)d_in[5];
    const float* w_ip   = (const float*)d_in[6];
    const float* w_ia   = (const float*)d_in[7];
    const float* w_is   = (const float*)d_in[8];
    const float* b_i    = (const float*)d_in[9];
    const float* w_op   = (const float*)d_in[10];
    const float* w_oa   = (const float*)d_in[11];
    const float* w_os   = (const float*)d_in[12];
    const float* b_o    = (const float*)d_in[13];
    const float* s_max  = (const float*)d_in[14];
    const float* s_fc   = (const float*)d_in[15];
    const float* mlp_w1 = (const float*)d_in[16];
    const float* mlp_b1 = (const float*)d_in[17];
    const float* mlp_w2 = (const float*)d_in[18];
    const float* mlp_b2 = (const float*)d_in[19];

    unsigned short* wpack  = (unsigned short*)d_ws;
    float* biasws          = (float*)((char*)d_ws + WP_BYTES);
    unsigned short* w2pack = (unsigned short*)((char*)d_ws + WP_BYTES + BIAS_BYTES);
    float* s_out = (float*)d_out;
    float* q_out = s_out + (size_t)B_TOT * L_SEQ * KDIM;

    int total = WP_ELEMS + NCOL + W2_ELEMS;
    DPL_prepack<<<(total + 255) / 256, 256, 0, stream>>>(
        w_rp, w_ra, w_rs, b_r, w_ip, w_ia, w_is, b_i,
        w_op, w_oa, w_os, b_o, mlp_w1, mlp_b1, mlp_w2,
        wpack, biasws, w2pack);

    DPL_scan<<<NBT, 1024, 0, stream>>>(
        prec, attr, s_max, s_fc, mlp_b2, wpack, biasws, w2pack, s_out, q_out);
}
